// Attention_2388001816996
// MI455X (gfx1250) — matrix operand order repaired and checked by reading, not compiled, not run on hardware
//
#include <hip/hip_runtime.h>

typedef __bf16 bf16;
typedef __attribute__((ext_vector_type(16))) __bf16 v16bf;
typedef __attribute__((ext_vector_type(8)))  __bf16 v8bf;
typedef __attribute__((ext_vector_type(4)))  __bf16 v4bf;
typedef __attribute__((ext_vector_type(8)))  float  v8f;

#define BATCH 4
#define SEQ   2048
#define CDIM  1024
#define NH    16
#define HD    64
#define QSCALE 0.125f   /* 64^-0.5 */

// ---------------------------------------------------------------------------
// Async global->LDS staging (GLOBAL_LOAD_ASYNC_TO_LDS_B128, ASYNCcnt).
// Probe-confirmed (round-2 diagnostic): param0 is AS1 (global) pointer to
// GCC-vector int4, so match that type exactly; param1 is the LDS address.
// ---------------------------------------------------------------------------
#if defined(__has_builtin)
#if __has_builtin(__builtin_amdgcn_global_load_async_to_lds_b128)
#define HAVE_ASYNC_LDS 1
#endif
#endif
#ifndef HAVE_ASYNC_LDS
#define HAVE_ASYNC_LDS 0
#endif

#if HAVE_ASYNC_LDS
typedef int v4i_gcc __attribute__((vector_size(16)));
typedef __attribute__((address_space(1))) v4i_gcc as1_v4i;
typedef __attribute__((address_space(3))) v4i_gcc as3_v4i;
#endif

static __device__ __forceinline__ void copy8_to_lds(bf16* dst_lds, const bf16* src) {
#if HAVE_ASYNC_LDS
  __builtin_amdgcn_global_load_async_to_lds_b128((as1_v4i*)src, (as3_v4i*)dst_lds, 0, 0);
#else
  *(uint4*)dst_lds = *(const uint4*)src;
#endif
}

static __device__ __forceinline__ void async_wait0() {
#if HAVE_ASYNC_LDS
#if __has_builtin(__builtin_amdgcn_s_wait_asynccnt)
  __builtin_amdgcn_s_wait_asynccnt(0);
#else
  asm volatile("s_wait_asynccnt 0x0" ::: "memory");
#endif
#endif
}

// ---------------------------------------------------------------------------
// WMMA fragment loaders (wave32; C/D layout per CDNA5 ISA §7.12.2)
// A-frag 16x32 bf16 (contiguous k, not the ISA table order; B matches): lanes 0-15 K 0-15; lanes 16-31 K 16-31
// B-frag 32x16 bf16 (col-major [col][K]): lane&15 = col, lane-half = K half
// C/D 16x16 f32: element r of lane L -> row = r + 8*(L>>4), col = L&15
// ---------------------------------------------------------------------------
static __device__ __forceinline__ v16bf load_a_frag(const bf16* base, int ld) {
  const int lane = threadIdx.x & 31;
  const int ln = lane & 15, hi = lane >> 4;
  const bf16* p = base + (size_t)ln * ld + hi * 16;
  return *(const v16bf*)p;
}

static __device__ __forceinline__ v16bf load_b_frag(const bf16* baseColMajor, int ld) {
  const int lane = threadIdx.x & 31;
  const int ln = lane & 15, hi = lane >> 4;
  const bf16* p = baseColMajor + (size_t)ln * ld + hi * 16;
  return *(const v16bf*)p;
}

static __device__ __forceinline__ v8f wmma_bf16(v16bf a, v16bf b, v8f c) {
  return __builtin_amdgcn_wmma_f32_16x16x32_bf16(false, a, false, b, (short)0, c,
                                                 false, false);
}

// ---------------------------------------------------------------------------
// fp32 -> bf16 conversion (one streaming pass)
// ---------------------------------------------------------------------------
__global__ void __launch_bounds__(256) cvt_f32_to_bf16(const float* __restrict__ in,
                                                       bf16* __restrict__ out, int n4) {
  int i = blockIdx.x * blockDim.x + threadIdx.x;
  if (i >= n4) return;
  float4 v = ((const float4*)in)[i];
  v4bf o = {(bf16)v.x, (bf16)v.y, (bf16)v.z, (bf16)v.w};
  ((v4bf*)out)[i] = o;
}

// ---------------------------------------------------------------------------
// Shared GEMM tile body: 128(M)x128(N) block tile, k-step 32, double-buffered
// LDS (async staged), 8 waves x (32x64) sub-tiles, 8 WMMAs per wave per step.
// ---------------------------------------------------------------------------
#define GEMM_BODY(X, W)                                                           \
  __shared__ __align__(32) bf16 Alds[2][128 * 32];                                \
  __shared__ __align__(32) bf16 Blds[2][128 * 32];                                \
  const int t = threadIdx.x;                                                      \
  const int wave = t >> 5, lane = t & 31, ln = lane & 15, hi = lane >> 4;         \
  const int wm = wave & 3, wn = wave >> 2;                                        \
  const size_t mbase = (size_t)blockIdx.y * 128;                                  \
  const size_t nbase = (size_t)blockIdx.x * 128;                                  \
  const v8f zf = {0.f, 0.f, 0.f, 0.f, 0.f, 0.f, 0.f, 0.f};                        \
  v8f acc[2][4];                                                                  \
  _Pragma("unroll") for (int i = 0; i < 2; ++i)                                   \
      _Pragma("unroll") for (int j = 0; j < 4; ++j) acc[i][j] = zf;               \
  const int srow = t >> 1, skh = (t & 1) << 4;                                    \
  const bf16* Arow = (X) + (mbase + srow) * CDIM + skh;                           \
  const bf16* Brow = (W) + (nbase + srow) * CDIM + skh;                           \
  /* prologue: stage k0=0 into buffer 0 */                                        \
  copy8_to_lds(&Alds[0][srow * 32 + skh], Arow);                                  \
  copy8_to_lds(&Alds[0][srow * 32 + skh + 8], Arow + 8);                          \
  copy8_to_lds(&Blds[0][srow * 32 + skh], Brow);                                  \
  copy8_to_lds(&Blds[0][srow * 32 + skh + 8], Brow + 8);                          \
  for (int k0 = 0; k0 < CDIM; k0 += 32) {                                         \
    const int cur = (k0 >> 5) & 1;                                                \
    async_wait0();                                                                \
    __syncthreads();                                                              \
    if (k0 + 32 < CDIM) { /* stage next tile into the other buffer */             \
      const int nb = cur ^ 1;                                                     \
      copy8_to_lds(&Alds[nb][srow * 32 + skh], Arow + k0 + 32);                   \
      copy8_to_lds(&Alds[nb][srow * 32 + skh + 8], Arow + k0 + 40);               \
      copy8_to_lds(&Blds[nb][srow * 32 + skh], Brow + k0 + 32);                   \
      copy8_to_lds(&Blds[nb][srow * 32 + skh + 8], Brow + k0 + 40);               \
      __builtin_prefetch(Arow + k0 + 64, 0, 1);                                   \
      __builtin_prefetch(Brow + k0 + 64, 0, 1);                                   \
    }                                                                             \
    v16bf a0 = load_a_frag(&Alds[cur][(wm * 32 + 0) * 32], 32);                   \
    v16bf a1 = load_a_frag(&Alds[cur][(wm * 32 + 16) * 32], 32);                  \
    v16bf b0 = load_b_frag(&Blds[cur][(wn * 64 + 0) * 32], 32);                   \
    v16bf b1 = load_b_frag(&Blds[cur][(wn * 64 + 16) * 32], 32);                  \
    v16bf b2 = load_b_frag(&Blds[cur][(wn * 64 + 32) * 32], 32);                  \
    v16bf b3 = load_b_frag(&Blds[cur][(wn * 64 + 48) * 32], 32);                  \
    acc[0][0] = wmma_bf16(a0, b0, acc[0][0]);                                     \
    acc[1][0] = wmma_bf16(a1, b0, acc[1][0]);                                     \
    acc[0][1] = wmma_bf16(a0, b1, acc[0][1]);                                     \
    acc[1][1] = wmma_bf16(a1, b1, acc[1][1]);                                     \
    acc[0][2] = wmma_bf16(a0, b2, acc[0][2]);                                     \
    acc[1][2] = wmma_bf16(a1, b2, acc[1][2]);                                     \
    acc[0][3] = wmma_bf16(a0, b3, acc[0][3]);                                     \
    acc[1][3] = wmma_bf16(a1, b3, acc[1][3]);                                     \
  }

// ---------------------------------------------------------------------------
// QKV GEMM: qkv[bn, d] = sum_c x[bn, c] * qkv_w[d, c]; epilogue scatters to
// per-head Q/K/V [B,H,N,hd] bf16, folding QSCALE into Q.
// ---------------------------------------------------------------------------
__global__ void __launch_bounds__(256) qkv_gemm(const bf16* __restrict__ X,
                                                const bf16* __restrict__ W,
                                                bf16* __restrict__ Qo,
                                                bf16* __restrict__ Ko,
                                                bf16* __restrict__ Vo) {
  GEMM_BODY(X, W)
#pragma unroll
  for (int ch = 0; ch < 2; ++ch)
#pragma unroll
    for (int cn = 0; cn < 4; ++cn)
#pragma unroll
      for (int r = 0; r < 8; ++r) {
        const int gm = (int)mbase + wm * 32 + ch * 16 + r + 8 * hi;
        const int gn = (int)nbase + wn * 64 + cn * 16 + ln;
        const float v = acc[ch][cn][r];
        const int b = gm >> 11, n = gm & (SEQ - 1);
        const int tsel = gn >> 10, h = (gn >> 6) & (NH - 1), e = gn & (HD - 1);
        const size_t idx = (((size_t)b * NH + h) * SEQ + n) * HD + e;
        if (tsel == 0)      Qo[idx] = (bf16)(v * QSCALE);
        else if (tsel == 1) Ko[idx] = (bf16)v;
        else                Vo[idx] = (bf16)v;
      }
}

// ---------------------------------------------------------------------------
// Output projection: out[bn, d] = sum_c O[bn, c] * proj_w[d, c] + proj_b[d]
// ---------------------------------------------------------------------------
__global__ void __launch_bounds__(256) proj_gemm(const bf16* __restrict__ X,
                                                 const bf16* __restrict__ W,
                                                 const float* __restrict__ bias,
                                                 float* __restrict__ out) {
  GEMM_BODY(X, W)
#pragma unroll
  for (int ch = 0; ch < 2; ++ch)
#pragma unroll
    for (int cn = 0; cn < 4; ++cn)
#pragma unroll
      for (int r = 0; r < 8; ++r) {
        const int gm = (int)mbase + wm * 32 + ch * 16 + r + 8 * hi;
        const int gn = (int)nbase + wn * 64 + cn * 16 + ln;
        out[(size_t)gm * CDIM + gn] = acc[ch][cn][r] + bias[gn];
      }
}

// ---------------------------------------------------------------------------
// Flash attention with softmax1 (extra zero logit): per (b,h), 128 query rows
// per block (16 rows/wave), key tiles of 32. Online softmax with m=0, l=1
// init implements the implicit zero logit exactly.
// ---------------------------------------------------------------------------
__global__ void __launch_bounds__(256) flash_attn(const bf16* __restrict__ Q,
                                                  const bf16* __restrict__ K,
                                                  const bf16* __restrict__ V,
                                                  const unsigned char* __restrict__ mask,
                                                  bf16* __restrict__ O) {
  __shared__ __align__(32) bf16 Klds[32 * 64];      // [key][d] row-major
  __shared__ __align__(32) bf16 Vtlds[64 * 32];     // [d][key] (B-frag col-major)
  __shared__ __align__(32) bf16 Plds[8][16 * 32];   // per-wave P tile
  const int t = threadIdx.x;
  const int wave = t >> 5, lane = t & 31, ln = lane & 15, hi = lane >> 4;
  const int b = blockIdx.z, h = blockIdx.y;
  const int q0 = blockIdx.x * 128;
  const size_t bh = (size_t)b * NH + h;

  const bf16* Qrow = Q + (bh * SEQ + q0 + wave * 16) * HD;
  const v16bf qf0 = load_a_frag(Qrow + 0, HD);
  const v16bf qf1 = load_a_frag(Qrow + 32, HD);

  const bf16* Kg = K + bh * SEQ * HD;
  const bf16* Vg = V + bh * SEQ * HD;
  const unsigned char* mrow = mask + (size_t)b * SEQ * SEQ;

  const v8f zf = {0.f, 0.f, 0.f, 0.f, 0.f, 0.f, 0.f, 0.f};
  v8f o[4];
  float m[8], l[8];
#pragma unroll
  for (int ct = 0; ct < 4; ++ct) o[ct] = zf;
#pragma unroll
  for (int r = 0; r < 8; ++r) { m[r] = 0.0f; l[r] = 1.0f; }  // softmax1 init

  for (int kt = 0; kt < SEQ / 32; ++kt) {
    __syncthreads();
    {  // stage 32 keys x 64 dims: K async to LDS, V transposed via VGPRs
      const int idx = t * 8;  // 256 threads * 8 bf16 = 2048 elements
      const bf16* ks = Kg + (size_t)kt * 32 * HD + idx;
      const bf16* vs = Vg + (size_t)kt * 32 * HD + idx;
      copy8_to_lds(&Klds[idx], ks);
      uint4 raw = *(const uint4*)vs;
      bf16 ve[8];
      __builtin_memcpy(ve, &raw, 16);
      const int vrow = idx >> 6;   // key index 0..31
      const int vcol = idx & 63;   // d
#pragma unroll
      for (int e = 0; e < 8; ++e) Vtlds[(vcol + e) * 32 + vrow] = ve[e];
      if (kt + 1 < SEQ / 32) {
        __builtin_prefetch(ks + 32 * HD, 0, 1);
        __builtin_prefetch(vs + 32 * HD, 0, 1);
      }
    }
    async_wait0();
    __syncthreads();

    // S = (Q*scale) K^T for 16 q-rows x 32 keys (scale folded into Q)
    v8f s0 = zf, s1 = zf;
    v16bf kb00 = load_b_frag(&Klds[0 * 64 + 0], 64);
    v16bf kb01 = load_b_frag(&Klds[0 * 64 + 32], 64);
    v16bf kb10 = load_b_frag(&Klds[16 * 64 + 0], 64);
    v16bf kb11 = load_b_frag(&Klds[16 * 64 + 32], 64);
    s0 = wmma_bf16(qf0, kb00, s0);
    s1 = wmma_bf16(qf0, kb10, s1);
    s0 = wmma_bf16(qf1, kb01, s0);
    s1 = wmma_bf16(qf1, kb11, s1);

    float esc[8];
#pragma unroll
    for (int r = 0; r < 8; ++r) {
      const int qg = q0 + wave * 16 + r + 8 * hi;
      const int k0i = kt * 32 + ln;
      float a0 = s0[r], a1 = s1[r];
      if (mrow[(size_t)qg * SEQ + k0i])      a0 = -3.0e38f;
      if (mrow[(size_t)qg * SEQ + k0i + 16]) a1 = -3.0e38f;
      float mx = fmaxf(a0, a1);
#pragma unroll
      for (int off = 1; off < 16; off <<= 1) mx = fmaxf(mx, __shfl_xor(mx, off));
      const float mn = fmaxf(m[r], mx);
      const float sc = __expf(m[r] - mn);
      const float p0 = __expf(a0 - mn);
      const float p1 = __expf(a1 - mn);
      float rs = p0 + p1;
#pragma unroll
      for (int off = 1; off < 16; off <<= 1) rs += __shfl_xor(rs, off);
      l[r] = l[r] * sc + rs;
      m[r] = mn;
      esc[r] = sc;
      // D-layout -> LDS so it can be re-read as an A-frag
      Plds[wave][(r + 8 * hi) * 32 + ln]      = (bf16)p0;
      Plds[wave][(r + 8 * hi) * 32 + 16 + ln] = (bf16)p1;
    }
    asm volatile("s_wait_dscnt 0x0" ::: "memory");  // within-wave LDS RAW
    const v16bf pf = load_a_frag(&Plds[wave][0], 32);
#pragma unroll
    for (int ct = 0; ct < 4; ++ct) {
      v8f co = o[ct];
#pragma unroll
      for (int r = 0; r < 8; ++r) co[r] *= esc[r];  // rescale old accumulator
      o[ct] = wmma_bf16(pf, load_b_frag(&Vtlds[ct * 16 * 32], 32), co);
    }
  }

  // normalize and store O in [B, N, H*hd] bf16 (ready for proj GEMM)
#pragma unroll
  for (int r = 0; r < 8; ++r) {
    const float inv = 1.0f / l[r];
    const int qg = q0 + wave * 16 + r + 8 * hi;
    bf16* orow = O + ((size_t)b * SEQ + qg) * CDIM + h * HD;
#pragma unroll
    for (int ct = 0; ct < 4; ++ct) orow[ct * 16 + ln] = (bf16)(o[ct][r] * inv);
  }
}

// ---------------------------------------------------------------------------
extern "C" void kernel_launch(void* const* d_in, const int* in_sizes, int n_in,
                              void* d_out, int out_size, void* d_ws, size_t ws_size,
                              hipStream_t stream) {
  const float* x = (const float*)d_in[0];
  const unsigned char* mask = (const unsigned char*)d_in[1];
  const float* qkv_w = (const float*)d_in[2];
  const float* proj_w = (const float*)d_in[3];
  const float* proj_b = (const float*)d_in[4];
  float* out = (float*)d_out;

  const size_t nX = (size_t)BATCH * SEQ * CDIM;   // 8,388,608
  const size_t nWqkv = (size_t)3 * CDIM * CDIM;   // 3,145,728
  const size_t nWproj = (size_t)CDIM * CDIM;      // 1,048,576

  bf16* p = (bf16*)d_ws;
  bf16* xb = p;     p += nX;
  bf16* wqkvb = p;  p += nWqkv;
  bf16* wprojb = p; p += nWproj;
  bf16* Qb = p;     p += nX;
  bf16* Kb = p;     p += nX;
  bf16* Vb = p;     p += nX;
  bf16* Ob = p;     p += nX;

  cvt_f32_to_bf16<<<(int)(nX / 4 / 256), 256, 0, stream>>>(x, xb, (int)(nX / 4));
  cvt_f32_to_bf16<<<(int)(nWqkv / 4 / 256), 256, 0, stream>>>(qkv_w, wqkvb, (int)(nWqkv / 4));
  cvt_f32_to_bf16<<<(int)(nWproj / 4 / 256), 256, 0, stream>>>(proj_w, wprojb, (int)(nWproj / 4));

  qkv_gemm<<<dim3(3 * CDIM / 128, BATCH * SEQ / 128), 256, 0, stream>>>(xb, wqkvb, Qb, Kb, Vb);
  flash_attn<<<dim3(SEQ / 128, NH, BATCH), 256, 0, stream>>>(Qb, Kb, Vb, mask, Ob);
  proj_gemm<<<dim3(CDIM / 128, BATCH * SEQ / 128), 256, 0, stream>>>(Ob, wprojb, proj_b, out);
}
